// Net_28896539967743
// MI455X (gfx1250) — matrix operand order repaired and checked by reading, not compiled, not run on hardware
//
#include <hip/hip_runtime.h>
#include <hip/hip_bf16.h>

typedef __bf16 bf16;
typedef __attribute__((ext_vector_type(16))) __bf16 v16bf;
typedef __attribute__((ext_vector_type(8)))  __bf16 v8bf;
typedef __attribute__((ext_vector_type(4)))  __bf16 v4bf;
typedef __attribute__((ext_vector_type(8)))  float  v8f;
typedef __attribute__((ext_vector_type(4)))  float  v4f;

#define M_RES   10000
#define EMBED   768
#define B_SZ    16
#define L_STEPS 12
#define NCLS    200
#define N_TILES (M_RES / 16)        // 625
#define K_CHUNKS 312                // 312*32 = 9984, tail covers K 9984..9999
#define K_TAIL  9984
#define WPB     8                   // waves per block (split-K factor); 312 = 8*39
#define CH_PER_WAVE 39

// ---------------- one-time conversion: Wres fp32 -> bf16 ----------------
__global__ void cvt_wres_k(const float* __restrict__ src, bf16* __restrict__ dst, long n4) {
    long tid    = (long)blockIdx.x * blockDim.x + threadIdx.x;
    long stride = (long)gridDim.x * blockDim.x;
    for (long idx = tid; idx < n4; idx += stride) {
        v4f f = *(const v4f*)(src + idx * 4);
        v4bf o;
        #pragma unroll
        for (int j = 0; j < 4; ++j) o[j] = (bf16)f[j];
        *(v4bf*)(dst + idx * 4) = o;
    }
}

// ---------------- init: zero h_bf16 ----------------
__global__ void init_k(bf16* __restrict__ h) {
    int tid = blockIdx.x * blockDim.x + threadIdx.x;
    if (tid < B_SZ * M_RES) h[tid] = (bf16)0.0f;
}

// ---------------- f_in = fs_t @ Win, stored [M, B] ----------------
__global__ void fin_k(const float* __restrict__ fs_t,   // [16, 768]
                      const float* __restrict__ Win,    // [768, 10000]
                      float* __restrict__ hpre) {       // [10000, 16]
    int tid = blockIdx.x * blockDim.x + threadIdx.x;
    if (tid >= M_RES * B_SZ) return;
    int i = tid >> 4, b = tid & 15;
    const float* f = fs_t + b * EMBED;
    const float* w = Win + i;
    float acc = 0.0f;
    #pragma unroll 4
    for (int e = 0; e < EMBED; ++e)
        acc += f[e] * w[(size_t)e * M_RES];
    hpre[tid] = acc;
}

// ---------------- core: hpre += Wres @ h.T  (WMMA bf16, fp32 accum, split-K x8) ----------------
__device__ inline v16bf cat16(v8bf lo, v8bf hi) {
    return __builtin_shufflevector(lo, hi, 0,1,2,3,4,5,6,7,8,9,10,11,12,13,14,15);
}

template<bool FROM_BF16>
__global__ void reservoir_wmma_k(const bf16*  __restrict__ WresB,  // [M, M] bf16
                                 const float* __restrict__ WresF,  // [M, M] fp32
                                 const bf16*  __restrict__ h,      // [16, M]
                                 float* __restrict__ hpre,         // [M, 16] in: f_in, out: h_new
                                 float* __restrict__ part)         // [N_TILES, 2] partial sums
{
    const int tile = blockIdx.x;                 // one 16-row tile per block
    const int wid  = threadIdx.x >> 5;           // 0..7 (split-K slice)
    const int lane = threadIdx.x & 31;
    const int i0   = tile * 16;
    const int half = lane >> 4;                  // 0 or 1
    const int l16  = lane & 15;

    v8f c;
    #pragma unroll
    for (int r = 0; r < 8; ++r) c[r] = 0.0f;

    const bf16*  aB = WresB + (size_t)(i0 + l16) * M_RES + half * 8;
    const float* aF = WresF + (size_t)(i0 + l16) * M_RES + half * 8;
    const bf16*  bP = h + (size_t)l16 * M_RES + half * 8;

    const int kbase = wid * (CH_PER_WAVE * 32);  // 1248 * wid

    #pragma unroll 4
    for (int it = 0; it < CH_PER_WAVE; ++it) {
        const int k0 = kbase + it * 32;
        v16bf a, bm;
        if (FROM_BF16) {
            v8bf a0 = *(const v8bf*)(aB + k0);
            v8bf a1 = *(const v8bf*)(aB + k0 + 16);
            a = cat16(a0, a1);
        } else {
            v4f f0 = *(const v4f*)(aF + k0);
            v4f f1 = *(const v4f*)(aF + k0 + 4);
            v4f f2 = *(const v4f*)(aF + k0 + 16);
            v4f f3 = *(const v4f*)(aF + k0 + 20);
            #pragma unroll
            for (int j = 0; j < 4; ++j) {
                a[j]      = (bf16)f0[j];
                a[4 + j]  = (bf16)f1[j];
                a[8 + j]  = (bf16)f2[j];
                a[12 + j] = (bf16)f3[j];
            }
        }
        {
            v8bf b0 = *(const v8bf*)(bP + k0);
            v8bf b1 = *(const v8bf*)(bP + k0 + 16);
            bm = cat16(b0, b1);
        }
        c = __builtin_amdgcn_wmma_f32_16x16x32_bf16(false, a, false, bm,
                                                    (short)0, c, false, false);
    }

    // tail (wave 7 only, wave-uniform branch): K = 9984..9999 valid
    if (wid == WPB - 1) {
        v16bf a, bm;
        if (FROM_BF16) {
            v8bf a0 = *(const v8bf*)(aB + K_TAIL);   // hi lanes end at elem 9999: in-bounds
            #pragma unroll
            for (int j = 0; j < 8; ++j) { a[j] = a0[j]; a[8 + j] = (bf16)0.0f; }
        } else {
            v4f f0 = *(const v4f*)(aF + K_TAIL);
            v4f f1 = *(const v4f*)(aF + K_TAIL + 4);
            #pragma unroll
            for (int j = 0; j < 4; ++j) {
                a[j]      = (bf16)f0[j];
                a[4 + j]  = (bf16)f1[j];
                a[8 + j]  = (bf16)0.0f;
                a[12 + j] = (bf16)0.0f;
            }
        }
        {
            v8bf b0 = *(const v8bf*)(bP + K_TAIL);
            #pragma unroll
            for (int j = 0; j < 8; ++j) { bm[j] = b0[j]; bm[8 + j] = (bf16)0.0f; }
        }
        c = __builtin_amdgcn_wmma_f32_16x16x32_bf16(false, a, false, bm,
                                                    (short)0, c, false, false);
    }

    // deterministic split-K combine via LDS
    __shared__ __align__(16) float lred[WPB - 1][32][8];
    if (wid > 0) {
        float* p = &lred[wid - 1][lane][0];
        #pragma unroll
        for (int r = 0; r < 8; ++r) p[r] = c[r];
    }
    __syncthreads();
    if (wid == 0) {
        #pragma unroll
        for (int w = 0; w < WPB - 1; ++w) {
            const float* p = &lred[w][lane][0];
            #pragma unroll
            for (int r = 0; r < 8; ++r) c[r] += p[r];
        }
        float s = 0.0f, sq = 0.0f;
        #pragma unroll
        for (int r = 0; r < 8; ++r) {
            float v = c[r] + hpre[(size_t)(i0 + r + half * 8) * 16 + l16]; // + f_in
            hpre[(size_t)(i0 + r + half * 8) * 16 + l16] = v;
            s += v; sq += v * v;
        }
        #pragma unroll
        for (int off = 16; off > 0; off >>= 1) {
            s  += __shfl_down(s,  off, 32);
            sq += __shfl_down(sq, off, 32);
        }
        if (lane == 0) { part[2 * tile] = s; part[2 * tile + 1] = sq; }
    }
}

// ---------------- deterministic reduce of per-tile partials ----------------
__global__ void reduce_k(const float* __restrict__ part, float* __restrict__ stats) {
    __shared__ float s1[256], s2[256];
    float a = 0.0f, b = 0.0f;
    for (int i = threadIdx.x; i < N_TILES; i += 256) {
        a += part[2 * i];
        b += part[2 * i + 1];
    }
    s1[threadIdx.x] = a; s2[threadIdx.x] = b;
    __syncthreads();
    for (int off = 128; off > 0; off >>= 1) {
        if ((int)threadIdx.x < off) {
            s1[threadIdx.x] += s1[threadIdx.x + off];
            s2[threadIdx.x] += s2[threadIdx.x + off];
        }
        __syncthreads();
    }
    if (threadIdx.x == 0) { stats[0] = s1[0]; stats[1] = s2[0]; }
}

// ---------------- layernorm (over all B*M) + leaky relu -> bf16 [16, M] ----------------
__global__ void normalize_k(const float* __restrict__ hpre,   // [M, 16]
                            const float* __restrict__ stats,
                            bf16* __restrict__ h) {           // [16, M]
    int tid = blockIdx.x * blockDim.x + threadIdx.x;
    if (tid >= M_RES * B_SZ) return;
    const float inv_n = 1.0f / (float)(M_RES * B_SZ);
    float mu  = stats[0] * inv_n;
    float var = stats[1] * inv_n - mu * mu;
    float sc  = rsqrtf(var + 1e-5f);
    int b = tid / M_RES, i = tid % M_RES;        // write-coalesced; strided reads hit L2
    float x = (hpre[(size_t)i * 16 + b] - mu) * sc;
    x = (x >= 0.0f) ? x : 0.01f * x;
    h[tid] = (bf16)x;
}

// ---------------- out = h @ Wout ----------------
__global__ void out_k(const bf16* __restrict__ h,      // [16, M]
                      const float* __restrict__ Wout,  // [M, 200]
                      float* __restrict__ out) {       // [16, 200]
    int tid = blockIdx.x * blockDim.x + threadIdx.x;
    if (tid >= B_SZ * NCLS) return;
    int b = tid / NCLS, cidx = tid % NCLS;
    const bf16*  hb = h + (size_t)b * M_RES;
    const float* w  = Wout + cidx;
    float acc = 0.0f;
    #pragma unroll 4
    for (int i = 0; i < M_RES; ++i)
        acc += (float)hb[i] * w[(size_t)i * NCLS];
    out[tid] = acc;
}

// ---------------- launch ----------------
extern "C" void kernel_launch(void* const* d_in, const int* in_sizes, int n_in,
                              void* d_out, int out_size, void* d_ws, size_t ws_size,
                              hipStream_t stream) {
    const float* fs   = (const float*)d_in[0];  // [12, 16, 768]
    const float* Win  = (const float*)d_in[1];  // [768, 10000]
    const float* Wres = (const float*)d_in[2];  // [10000, 10000]
    const float* Wout = (const float*)d_in[3];  // [10000, 200]
    float* out = (float*)d_out;

    char* ws = (char*)d_ws;
    // layout: stats(2f @0) | partials(625*2f @128) | h_bf16 @5632 | hpre @325632 | WresB @965632
    float* stats = (float*)(ws + 0);
    float* part  = (float*)(ws + 128);
    bf16*  hB    = (bf16*)(ws + 5632);
    float* hpre  = (float*)(ws + 325632);
    bf16*  WresB = (bf16*)(ws + 965632);
    const size_t needB = 965632ull + (size_t)M_RES * M_RES * 2ull;
    const bool useB = (ws_size >= needB);

    init_k<<<N_TILES, 256, 0, stream>>>(hB);
    if (useB)
        cvt_wres_k<<<4096, 256, 0, stream>>>(Wres, WresB, (long)M_RES * M_RES / 4);

    for (int t = 0; t < L_STEPS; ++t) {
        fin_k<<<N_TILES, 256, 0, stream>>>(fs + (size_t)t * B_SZ * EMBED, Win, hpre);
        if (useB)
            reservoir_wmma_k<true><<<N_TILES, 32 * WPB, 0, stream>>>(WresB, Wres, hB, hpre, part);
        else
            reservoir_wmma_k<false><<<N_TILES, 32 * WPB, 0, stream>>>(WresB, Wres, hB, hpre, part);
        reduce_k<<<1, 256, 0, stream>>>(part, stats);
        normalize_k<<<N_TILES, 256, 0, stream>>>(hpre, stats, hB);
    }
    out_k<<<(B_SZ * NCLS + 255) / 256, 256, 0, stream>>>(hB, Wout, out);
}
